// SmallOps_53678501265850
// MI455X (gfx1250) — 8-bit matrix operand order repaired and checked by reading, compiled, not run on hardware
//
#include <hip/hip_runtime.h>
#include <hip/hip_bf16.h>
#include <stdint.h>

#define T_      512
#define K_      8
#define H_      1024
#define I_      512
#define E_      64
#define M_      (T_ * K_)
#define TWO_I_  (2 * I_)
#define MAXTILES_ 320

typedef __attribute__((ext_vector_type(8))) int v8i;

__device__ __forceinline__ int quant_byte(float v, float sc) {
  float r = rintf(v / sc);
  r = fminf(fmaxf(r, -128.0f), 127.0f);
  return ((int)r) & 0xff;
}

__global__ __launch_bounds__(256) void moe_w_to_i8_t(
    const float* __restrict__ src, signed char* __restrict__ dst, int Kd, int N) {
  __shared__ float tile[32][33];
  const int e  = blockIdx.z;
  const int k0 = blockIdx.x * 32;
  const int n0 = blockIdx.y * 32;
  const int tid = threadIdx.x;
  const float* s = src + (size_t)e * Kd * N;
#pragma unroll
  for (int i = 0; i < 4; ++i) {
    int idx = tid + i * 256;
    int kk = idx >> 5, nn = idx & 31;
    tile[kk][nn] = s[(size_t)(k0 + kk) * N + (n0 + nn)];
  }
  __syncthreads();
  int nn = tid >> 3, dw = tid & 7;
  int b0 = ((int)tile[dw * 4 + 0][nn]) & 0xff;
  int b1 = ((int)tile[dw * 4 + 1][nn]) & 0xff;
  int b2 = ((int)tile[dw * 4 + 2][nn]) & 0xff;
  int b3 = ((int)tile[dw * 4 + 3][nn]) & 0xff;
  int packed = b0 | (b1 << 8) | (b2 << 16) | (b3 << 24);
  signed char* dbase = dst + (size_t)e * N * Kd;
  *(int*)(dbase + (size_t)(n0 + nn) * Kd + k0 + dw * 4) = packed;
}

__global__ __launch_bounds__(256) void moe_dispatch(
    const int* __restrict__ ids,
    int* __restrict__ row_tok, int* __restrict__ inv,
    int* __restrict__ tileExpert, int* __restrict__ tileRow,
    int* __restrict__ tileEnd, int* __restrict__ numTiles) {
  __shared__ int s_cnt[E_], s_off[E_], s_cur[E_];
  const int tid = threadIdx.x;
  if (tid < E_) { s_cnt[tid] = 0; s_cur[tid] = 0; }
  __syncthreads();
  for (int m = tid; m < M_; m += 256) atomicAdd(&s_cnt[ids[m]], 1);
  __syncthreads();
  if (tid == 0) {
    int acc = 0, nt = 0;
    for (int e = 0; e < E_; ++e) {
      s_off[e] = acc;
      int c = s_cnt[e];
      for (int r = 0; r < c; r += 16) {
        tileExpert[nt] = e; tileRow[nt] = acc + r; tileEnd[nt] = acc + c; ++nt;
      }
      acc += c;
    }
    *numTiles = nt;
  }
  __syncthreads();
  for (int m = tid; m < M_; m += 256) {
    int e = ids[m];
    int pos = s_off[e] + atomicAdd(&s_cur[e], 1);
    row_tok[pos] = m / K_;
    inv[m] = pos;
  }
}

__global__ __launch_bounds__(256) void moe_quant_x(
    const float* __restrict__ x, const int* __restrict__ row_tok,
    signed char* __restrict__ q, float* __restrict__ dyn) {
  const int row = blockIdx.x, tid = threadIdx.x;
  const int tok = row_tok[row];
  const float4 v = ((const float4*)(x + (size_t)tok * H_))[tid];
  float m = fmaxf(fmaxf(fabsf(v.x), fabsf(v.y)), fmaxf(fabsf(v.z), fabsf(v.w)));
  __shared__ float red[256];
  red[tid] = m; __syncthreads();
  for (int s = 128; s > 0; s >>= 1) {
    if (tid < s) red[tid] = fmaxf(red[tid], red[tid + s]);
    __syncthreads();
  }
  const float sc = fmaxf(red[0] * (1.0f / 127.0f), 1e-9f);
  if (tid == 0) dyn[row] = sc;
  int packed = quant_byte(v.x, sc) | (quant_byte(v.y, sc) << 8) |
               (quant_byte(v.z, sc) << 16) | (quant_byte(v.w, sc) << 24);
  ((int*)q)[(size_t)row * (H_ / 4) + tid] = packed;
}

template <int Kd, int N>
__global__ __launch_bounds__(128) void moe_gemm_i8(
    const signed char* __restrict__ Aq, const signed char* __restrict__ Bq,
    const float* __restrict__ rowScale, const float* __restrict__ colScale,
    float* __restrict__ Out,
    const int* __restrict__ tileExpert, const int* __restrict__ tileRow,
    const int* __restrict__ tileEnd, const int* __restrict__ numTilesPtr) {
  const int t = blockIdx.y;
  if (t >= *numTilesPtr) return;
  const int e    = tileExpert[t];
  const int r0   = tileRow[t];
  const int rend = tileEnd[t];
  const int wave = threadIdx.x >> 5;
  const int lane = threadIdx.x & 31;
  const int lo = lane & 15, hi = lane >> 4;

  const int col0 = blockIdx.x * 128 + wave * 32 + lo;
  int arow = r0 + lo;
  if (arow >= rend) arow = rend - 1;

  const signed char* aptr = Aq + (size_t)arow * Kd + hi * 16;
  const signed char* bptr0 = Bq + ((size_t)e * N + col0) * Kd + hi * 16;
  const signed char* bptr1 = bptr0 + (size_t)16 * Kd;

  v8i acc0 = {0, 0, 0, 0, 0, 0, 0, 0};
  v8i acc1 = {0, 0, 0, 0, 0, 0, 0, 0};
#pragma unroll 4
  for (int k0 = 0; k0 < Kd; k0 += 64) {
    __builtin_prefetch(bptr0 + k0 + 512, 0, 1);
    __builtin_prefetch(bptr1 + k0 + 512, 0, 1);
    int4 a03 = *(const int4*)(aptr + k0);
    int4 a47 = *(const int4*)(aptr + k0 + 32);
    v8i a = {a03.x, a03.y, a03.z, a03.w, a47.x, a47.y, a47.z, a47.w};
    int4 p03 = *(const int4*)(bptr0 + k0);
    int4 p47 = *(const int4*)(bptr0 + k0 + 32);
    int4 q03 = *(const int4*)(bptr1 + k0);
    int4 q47 = *(const int4*)(bptr1 + k0 + 32);
    v8i b0 = {p03.x, p03.y, p03.z, p03.w, p47.x, p47.y, p47.z, p47.w};
    v8i b1 = {q03.x, q03.y, q03.z, q03.w, q47.x, q47.y, q47.z, q47.w};
    acc0 = __builtin_amdgcn_wmma_i32_16x16x64_iu8(true, a, true, b0, acc0,
                                                  false, false);
    acc1 = __builtin_amdgcn_wmma_i32_16x16x64_iu8(true, a, true, b1, acc1,
                                                  false, false);
  }
  const float cs0 = colScale[(size_t)e * N + col0];
  const float cs1 = colScale[(size_t)e * N + col0 + 16];
  const int baseRow = r0 + hi * 8;
#pragma unroll
  for (int r = 0; r < 8; ++r) {
    int row = baseRow + r;
    if (row < rend) {
      float rs = rowScale[row];
      Out[(size_t)row * N + col0]      = (float)acc0[r] * rs * cs0;
      Out[(size_t)row * N + col0 + 16] = (float)acc1[r] * rs * cs1;
    }
  }
}

__global__ __launch_bounds__(128) void moe_swiglu_quant(
    const float* __restrict__ y1, signed char* __restrict__ hq,
    float* __restrict__ hs) {
  const int row = blockIdx.x, tid = threadIdx.x;
  const float4 a = ((const float4*)(y1 + (size_t)row * TWO_I_))[tid];
  const float4 b = ((const float4*)(y1 + (size_t)row * TWO_I_ + I_))[tid];
  float h0 = (a.x / (1.0f + __expf(-a.x))) * b.x;
  float h1 = (a.y / (1.0f + __expf(-a.y))) * b.y;
  float h2 = (a.z / (1.0f + __expf(-a.z))) * b.z;
  float h3 = (a.w / (1.0f + __expf(-a.w))) * b.w;
  float m = fmaxf(fmaxf(fabsf(h0), fabsf(h1)), fmaxf(fabsf(h2), fabsf(h3)));
  __shared__ float red[128];
  red[tid] = m; __syncthreads();
  for (int s = 64; s > 0; s >>= 1) {
    if (tid < s) red[tid] = fmaxf(red[tid], red[tid + s]);
    __syncthreads();
  }
  const float sc = fmaxf(red[0] * (1.0f / 127.0f), 1e-9f);
  if (tid == 0) hs[row] = sc;
  int packed = quant_byte(h0, sc) | (quant_byte(h1, sc) << 8) |
               (quant_byte(h2, sc) << 16) | (quant_byte(h3, sc) << 24);
  ((int*)hq)[(size_t)row * (I_ / 4) + tid] = packed;
}

__global__ __launch_bounds__(256) void moe_combine(
    const float* __restrict__ y2, const int* __restrict__ inv,
    const float* __restrict__ es, float* __restrict__ out) {
  const int t = blockIdx.x, tid = threadIdx.x;
  __shared__ int rows[K_];
  __shared__ float w[K_];
  if (tid < K_) { rows[tid] = inv[t * K_ + tid]; w[tid] = es[t * K_ + tid]; }
  __syncthreads();
  float4 acc = {0.f, 0.f, 0.f, 0.f};
#pragma unroll
  for (int k = 0; k < K_; ++k) {
    const float4 v = ((const float4*)(y2 + (size_t)rows[k] * H_))[tid];
    const float wk = w[k];
    acc.x += v.x * wk; acc.y += v.y * wk; acc.z += v.z * wk; acc.w += v.w * wk;
  }
  ((float4*)(out + (size_t)t * H_))[tid] = acc;
}

extern "C" void kernel_launch(void* const* d_in, const int* in_sizes, int n_in,
                              void* d_out, int out_size, void* d_ws,
                              size_t ws_size, hipStream_t stream) {
  const float* x   = (const float*)d_in[0];
  const int*   ids = (const int*)d_in[1];
  const float* w1  = (const float*)d_in[2];
  const float* w1s = (const float*)d_in[3];
  const float* w2  = (const float*)d_in[4];
  const float* w2s = (const float*)d_in[5];
  const float* es  = (const float*)d_in[7];
  float* out = (float*)d_out;

  uint8_t* ws = (uint8_t*)d_ws;
  size_t o = 0;
  auto take = [&](size_t bytes) {
    uint8_t* p = ws + o;
    o = (o + bytes + 255) & ~(size_t)255;
    return p;
  };
  signed char* w1q = (signed char*)take((size_t)E_ * TWO_I_ * H_);
  signed char* w2q = (signed char*)take((size_t)E_ * H_ * I_);
  signed char* q   = (signed char*)take((size_t)M_ * H_);
  signed char* hq  = (signed char*)take((size_t)M_ * I_);
  float* y1  = (float*)take((size_t)M_ * TWO_I_ * 4);
  float* y2  = (float*)take((size_t)M_ * H_ * 4);
  float* dyn = (float*)take((size_t)M_ * 4);
  float* hsc = (float*)take((size_t)M_ * 4);
  int* row_tok    = (int*)take((size_t)M_ * 4);
  int* inv        = (int*)take((size_t)M_ * 4);
  int* tileExpert = (int*)take((size_t)MAXTILES_ * 4);
  int* tileRow    = (int*)take((size_t)MAXTILES_ * 4);
  int* tileEnd    = (int*)take((size_t)MAXTILES_ * 4);
  int* numTiles   = (int*)take(256);

  moe_w_to_i8_t<<<dim3(H_ / 32, TWO_I_ / 32, E_), 256, 0, stream>>>(w1, w1q, H_, TWO_I_);
  moe_w_to_i8_t<<<dim3(I_ / 32, H_ / 32, E_), 256, 0, stream>>>(w2, w2q, I_, H_);

  moe_dispatch<<<1, 256, 0, stream>>>(ids, row_tok, inv, tileExpert, tileRow,
                                      tileEnd, numTiles);
  moe_quant_x<<<M_, 256, 0, stream>>>(x, row_tok, q, dyn);

  moe_gemm_i8<H_, TWO_I_><<<dim3(TWO_I_ / 128, MAXTILES_), 128, 0, stream>>>(
      q, w1q, dyn, w1s, y1, tileExpert, tileRow, tileEnd, numTiles);

  moe_swiglu_quant<<<M_, 128, 0, stream>>>(y1, hq, hsc);

  moe_gemm_i8<I_, H_><<<dim3(H_ / 128, MAXTILES_), 128, 0, stream>>>(
      hq, w2q, hsc, w2s, y2, tileExpert, tileRow, tileEnd, numTiles);

  moe_combine<<<T_, 256, 0, stream>>>(y2, inv, es, out);
}
